// AttenModule_38826504356622
// MI455X (gfx1250) — compile-verified, matrix operand order checked by reading, not run on hardware
//
#include <hip/hip_runtime.h>
#include <hip/hip_bf16.h>
#include <stdint.h>

// Problem constants (from reference)
#define B_  64
#define N_  1024
#define M_  80
#define C1_ 288
#define D_  256

typedef __attribute__((ext_vector_type(16))) __bf16 v16bf;
typedef __attribute__((ext_vector_type(8)))  float  v8f;

#define ZERO8 {0.f,0.f,0.f,0.f,0.f,0.f,0.f,0.f}
#define STG_STRIDE 36   // floats; 16B-aligned rows, conflict-free (36*r mod 64 distinct)

union BF16Frag {
  v16bf v;
  unsigned short u[16];
  uint4 q[2];
};

__device__ __forceinline__ unsigned short f2bf(float f) {
  unsigned int x = __float_as_uint(f);
  x += 0x7fffu + ((x >> 16) & 1u);   // round-to-nearest-even truncation
  return (unsigned short)(x >> 16);
}
__device__ __forceinline__ float bf2f(unsigned short s) {
  return __uint_as_float(((unsigned int)s) << 16);
}
// 16-bit A-matrix 16x32 element->K mapping (ISA 7.12.2), mirrored for B frags
__device__ __forceinline__ int klocal(int g, int e) {
  return (e < 8) ? (g * 8 + e) : (16 + g * 8 + (e - 8));
}
__device__ __forceinline__ void lds_fence() {
  asm volatile("s_wait_dscnt 0x0" ::: "memory");
}

// ---------------- prep: pack weight (K x 256, row-major f32) into B-fragment order ----------
__global__ void pack_weight_kernel(const float* __restrict__ src,
                                   unsigned short* __restrict__ dst,
                                   int nkc) {
  int t = blockIdx.x * blockDim.x + threadIdx.x;
  int total = nkc * 16 * 32;
  if (t >= total) return;
  int lane = t & 31;
  int fid  = t >> 5;          // kc*16 + cn
  int cn = fid & 15;
  int kc = fid >> 4;
  int g = lane >> 4, r = lane & 15;
  int n = cn * 16 + r;
  unsigned short* o = dst + ((size_t)fid * 32 + lane) * 16;
#pragma unroll
  for (int e = 0; e < 16; ++e) {
    int k = kc * 32 + klocal(g, e);
    o[e] = f2bf(src[(size_t)k * D_ + n]);
  }
}

// ---------------- prep: pack lang_feat (B x 80 x 256) into B-fragments, K padded to 96 -------
__global__ void pack_lang_kernel(const float* __restrict__ L,
                                 unsigned short* __restrict__ dst) {
  int t = blockIdx.x * blockDim.x + threadIdx.x;
  int total = B_ * 3 * 16 * 32;
  if (t >= total) return;
  int lane = t & 31;
  int fid  = t >> 5;          // b*48 + kc*16 + cn
  int cn = fid & 15;
  int kc = (fid >> 4) % 3;
  int bb = fid / 48;
  int g = lane >> 4, r = lane & 15;
  int n = cn * 16 + r;
  unsigned short* o = dst + ((size_t)fid * 32 + lane) * 16;
#pragma unroll
  for (int e = 0; e < 16; ++e) {
    int k = kc * 32 + klocal(g, e);
    float v = (k < M_) ? L[((size_t)bb * M_ + k) * D_ + n] : 0.0f;
    o[e] = f2bf(v);
  }
}

// ---------------- prep: masked language logits  s_l[b,m] = lang_feat[b,m,:]·w_l + ba ---------
__global__ void lang_score_kernel(const float* __restrict__ L,
                                  const int* __restrict__ mask,
                                  const float* __restrict__ Wa,
                                  const float* __restrict__ ba,
                                  float* __restrict__ slm) {
  int t = blockIdx.x * blockDim.x + threadIdx.x;
  if (t >= B_ * M_) return;
  int b = t / M_, m = t % M_;
  const float* row = L + ((size_t)b * M_ + m) * D_;
  const float* wl = Wa + D_;      // w_l = Wa[D:, 0]
  float acc = 0.f;
  for (int d = 0; d < D_; ++d) acc += row[d] * wl[d];
  slm[t] = (mask[t] != 0) ? (acc + ba[0]) : -1.0e30f;
}

// ---------------- hidden = relu(X @ W + b) repacked to bf16 A-fragments ----------------------
// Two independent column-chunk accumulator chains per j so B-fragment loads overlap WMMAs.
__device__ __forceinline__ void hidden_gemm(const BF16Frag (&Xf)[9],
                                            const unsigned short* __restrict__ Wp,
                                            const float* __restrict__ bias,
                                            float* __restrict__ stg,
                                            int lane, int g, int r,
                                            BF16Frag (&hf)[8]) {
#pragma unroll
  for (int j = 0; j < 8; ++j) {           // pairs of 16-col chunks -> one K=32 A-fragment
    const int cn0 = 2 * j, cn1 = 2 * j + 1;
    v8f acc0 = ZERO8, acc1 = ZERO8;
#pragma unroll
    for (int kc = 0; kc < 9; ++kc) {
      v16bf bf0 = *reinterpret_cast<const v16bf*>(
          Wp + ((size_t)(kc * 16 + cn0) * 32 + lane) * 16);
      v16bf bf1 = *reinterpret_cast<const v16bf*>(
          Wp + ((size_t)(kc * 16 + cn1) * 32 + lane) * 16);
      acc0 = __builtin_amdgcn_wmma_f32_16x16x32_bf16(
          false, Xf[kc].v, false, bf0, (short)0, acc0, false, false);
      acc1 = __builtin_amdgcn_wmma_f32_16x16x32_bf16(
          false, Xf[kc].v, false, bf1, (short)0, acc1, false, false);
    }
    const float bv0 = bias[cn0 * 16 + r];
    const float bv1 = bias[cn1 * 16 + r];
#pragma unroll
    for (int v = 0; v < 8; ++v) {          // C-layout -> LDS (lane=col, elem=row)
      stg[(v + g * 8) * STG_STRIDE + r]      = fmaxf(acc0[v] + bv0, 0.0f);
      stg[(v + g * 8) * STG_STRIDE + 16 + r] = fmaxf(acc1[v] + bv1, 0.0f);
    }
    lds_fence();
    // read back in A-layout (lane=row): contiguous groups of 8 -> 4x ds_load_b128
    const float4* s0 = reinterpret_cast<const float4*>(stg + r * STG_STRIDE + g * 8);
    const float4* s1 = reinterpret_cast<const float4*>(stg + r * STG_STRIDE + 16 + g * 8);
    float4 f0 = s0[0], f1 = s0[1], f2 = s1[0], f3 = s1[1];
    hf[j].u[0]  = f2bf(f0.x); hf[j].u[1]  = f2bf(f0.y);
    hf[j].u[2]  = f2bf(f0.z); hf[j].u[3]  = f2bf(f0.w);
    hf[j].u[4]  = f2bf(f1.x); hf[j].u[5]  = f2bf(f1.y);
    hf[j].u[6]  = f2bf(f1.z); hf[j].u[7]  = f2bf(f1.w);
    hf[j].u[8]  = f2bf(f2.x); hf[j].u[9]  = f2bf(f2.y);
    hf[j].u[10] = f2bf(f2.z); hf[j].u[11] = f2bf(f2.w);
    hf[j].u[12] = f2bf(f3.x); hf[j].u[13] = f2bf(f3.y);
    hf[j].u[14] = f2bf(f3.z); hf[j].u[15] = f2bf(f3.w);
    lds_fence();
  }
}

// ---------------- fused main kernel: one wave per 16-row object tile -------------------------
#define WAVES 8
__global__ __launch_bounds__(256) void fused_atten_kernel(
    const float* __restrict__ X,
    const float* __restrict__ slm,
    const unsigned short* __restrict__ W1p,
    const unsigned short* __restrict__ W2p,
    const unsigned short* __restrict__ W3p,
    const unsigned short* __restrict__ W4p,
    const unsigned short* __restrict__ Lp,
    const float* __restrict__ b1, const float* __restrict__ b2,
    const float* __restrict__ b3, const float* __restrict__ b4,
    const float* __restrict__ Wa,
    const float* __restrict__ Wsv, const float* __restrict__ bs,
    float* __restrict__ out) {
  __shared__ float stage[WAVES][16 * STG_STRIDE];
  __shared__ float soLDS[WAVES][16];

  const int wv   = threadIdx.x >> 5;
  const int lane = threadIdx.x & 31;
  const int g = lane >> 4, r = lane & 15;
  const int tile = blockIdx.x * WAVES + wv;   // 0..4095
  const int bIdx = tile >> 6;                 // / (N/16)
  const int row0 = (tile & 63) << 4;
  float* stg = &stage[wv][0];

  // ---- load 16x288 X tile as 9 bf16 A-fragments (contiguous float4 loads) ----
  BF16Frag Xf[9];
  const float* Xrow = X + ((size_t)(bIdx * N_ + row0 + r)) * C1_;
#pragma unroll
  for (int kc = 0; kc < 9; ++kc) {
    const float4* p0 = reinterpret_cast<const float4*>(Xrow + kc * 32 + g * 8);
    const float4* p1 = reinterpret_cast<const float4*>(Xrow + kc * 32 + 16 + g * 8);
    float4 a0 = p0[0], a1 = p0[1], a2 = p1[0], a3 = p1[1];
    Xf[kc].u[0]  = f2bf(a0.x); Xf[kc].u[1]  = f2bf(a0.y);
    Xf[kc].u[2]  = f2bf(a0.z); Xf[kc].u[3]  = f2bf(a0.w);
    Xf[kc].u[4]  = f2bf(a1.x); Xf[kc].u[5]  = f2bf(a1.y);
    Xf[kc].u[6]  = f2bf(a1.z); Xf[kc].u[7]  = f2bf(a1.w);
    Xf[kc].u[8]  = f2bf(a2.x); Xf[kc].u[9]  = f2bf(a2.y);
    Xf[kc].u[10] = f2bf(a2.z); Xf[kc].u[11] = f2bf(a2.w);
    Xf[kc].u[12] = f2bf(a3.x); Xf[kc].u[13] = f2bf(a3.y);
    Xf[kc].u[14] = f2bf(a3.z); Xf[kc].u[15] = f2bf(a3.w);
  }

  // ---- GEMM1 / GEMM3: hidden states as bf16 A-fragments ----
  BF16Frag h1f[8], h3f[8];
  hidden_gemm(Xf, W1p, b1, stg, lane, g, r, h1f);
  hidden_gemm(Xf, W3p, b3, stg, lane, g, r, h3f);

  // ---- GEMM2: oa = h1 @ W2 + b2, folded into s_o[row] = oa·w_o (dual chains) ----
  float sop[8] = {0.f,0.f,0.f,0.f,0.f,0.f,0.f,0.f};
  for (int cn = 0; cn < 16; cn += 2) {
    v8f a0 = ZERO8, a1 = ZERO8;
#pragma unroll
    for (int kc = 0; kc < 8; ++kc) {
      v16bf bf0 = *reinterpret_cast<const v16bf*>(
          W2p + ((size_t)(kc * 16 + cn) * 32 + lane) * 16);
      v16bf bf1 = *reinterpret_cast<const v16bf*>(
          W2p + ((size_t)(kc * 16 + cn + 1) * 32 + lane) * 16);
      a0 = __builtin_amdgcn_wmma_f32_16x16x32_bf16(
          false, h1f[kc].v, false, bf0, (short)0, a0, false, false);
      a1 = __builtin_amdgcn_wmma_f32_16x16x32_bf16(
          false, h1f[kc].v, false, bf1, (short)0, a1, false, false);
    }
    const float bv0 = b2[cn * 16 + r],       wo0 = Wa[cn * 16 + r];
    const float bv1 = b2[(cn + 1) * 16 + r], wo1 = Wa[(cn + 1) * 16 + r];
#pragma unroll
    for (int v = 0; v < 8; ++v)
      sop[v] += (a0[v] + bv0) * wo0 + (a1[v] + bv1) * wo1;
  }
#pragma unroll
  for (int v = 0; v < 8; ++v) {                // sum over the 16 columns held per half-wave
    sop[v] += __shfl_xor(sop[v], 1, 32);
    sop[v] += __shfl_xor(sop[v], 2, 32);
    sop[v] += __shfl_xor(sop[v], 4, 32);
    sop[v] += __shfl_xor(sop[v], 8, 32);
  }
  if (r == 0) {
#pragma unroll
    for (int v = 0; v < 8; ++v) soLDS[wv][g * 8 + v] = sop[v];
  }
  lds_fence();
  const float so_row = soLDS[wv][r];

  // ---- masked softmax over M=80, packed straight into bf16 A-fragments ----
  const float* slmb = slm + bIdx * M_;
  float mx = -3.0e38f;
#pragma unroll
  for (int kc = 0; kc < 3; ++kc)
#pragma unroll
    for (int e = 0; e < 16; ++e) {
      int k = kc * 32 + klocal(g, e);
      float sc = (k < M_) ? (so_row + slmb[k]) : -3.0e38f;
      mx = fmaxf(mx, sc);
    }
  mx = fmaxf(mx, __shfl_xor(mx, 16, 32));      // rows split across lane and lane^16
  BF16Frag attA[3];
  float sum = 0.f;
#pragma unroll
  for (int kc = 0; kc < 3; ++kc)
#pragma unroll
    for (int e = 0; e < 16; ++e) {
      int k = kc * 32 + klocal(g, e);
      float ex = 0.f;
      if (k < M_) ex = __expf(so_row + slmb[k] - mx);
      attA[kc].u[e] = f2bf(ex);
      sum += ex;
    }
  sum += __shfl_xor(sum, 16, 32);
  const float inv = 1.0f / sum;
#pragma unroll
  for (int kc = 0; kc < 3; ++kc)
#pragma unroll
    for (int e = 0; e < 16; ++e)
      attA[kc].u[e] = f2bf(bf2f(attA[kc].u[e]) * inv);

  // ---- fused column loop: osc (GEMM4) + att_feat (att @ lang), 4 chains in flight ----
  float S1[8] = {0.f,0.f,0.f,0.f,0.f,0.f,0.f,0.f};
  float S2[8] = {0.f,0.f,0.f,0.f,0.f,0.f,0.f,0.f};
  const unsigned short* Lpb = Lp + (size_t)bIdx * 48 * 32 * 16;
  for (int cn = 0; cn < 16; cn += 2) {
    v8f o0 = ZERO8, o1 = ZERO8;
#pragma unroll
    for (int kc = 0; kc < 8; ++kc) {
      v16bf bf0 = *reinterpret_cast<const v16bf*>(
          W4p + ((size_t)(kc * 16 + cn) * 32 + lane) * 16);
      v16bf bf1 = *reinterpret_cast<const v16bf*>(
          W4p + ((size_t)(kc * 16 + cn + 1) * 32 + lane) * 16);
      o0 = __builtin_amdgcn_wmma_f32_16x16x32_bf16(
          false, h3f[kc].v, false, bf0, (short)0, o0, false, false);
      o1 = __builtin_amdgcn_wmma_f32_16x16x32_bf16(
          false, h3f[kc].v, false, bf1, (short)0, o1, false, false);
    }
    v8f af0 = ZERO8, af1 = ZERO8;
#pragma unroll
    for (int kc = 0; kc < 3; ++kc) {
      v16bf bf0 = *reinterpret_cast<const v16bf*>(
          Lpb + ((size_t)(kc * 16 + cn) * 32 + lane) * 16);
      v16bf bf1 = *reinterpret_cast<const v16bf*>(
          Lpb + ((size_t)(kc * 16 + cn + 1) * 32 + lane) * 16);
      af0 = __builtin_amdgcn_wmma_f32_16x16x32_bf16(
          false, attA[kc].v, false, bf0, (short)0, af0, false, false);
      af1 = __builtin_amdgcn_wmma_f32_16x16x32_bf16(
          false, attA[kc].v, false, bf1, (short)0, af1, false, false);
    }
    const float bv0  = b4[cn * 16 + r],       wsn0 = Wsv[cn * 16 + r];
    const float bv1  = b4[(cn + 1) * 16 + r], wsn1 = Wsv[(cn + 1) * 16 + r];
#pragma unroll
    for (int v = 0; v < 8; ++v) {
      float x0 = fmaxf(af0[v] * (o0[v] + bv0), 0.f);
      float x1 = fmaxf(af1[v] * (o1[v] + bv1), 0.f);
      S1[v] += x0 * x0 + x1 * x1;
      S2[v] += x0 * wsn0 + x1 * wsn1;
    }
  }
#pragma unroll
  for (int v = 0; v < 8; ++v) {
    S1[v] += __shfl_xor(S1[v], 1, 32); S1[v] += __shfl_xor(S1[v], 2, 32);
    S1[v] += __shfl_xor(S1[v], 4, 32); S1[v] += __shfl_xor(S1[v], 8, 32);
    S2[v] += __shfl_xor(S2[v], 1, 32); S2[v] += __shfl_xor(S2[v], 2, 32);
    S2[v] += __shfl_xor(S2[v], 4, 32); S2[v] += __shfl_xor(S2[v], 8, 32);
  }
  if (r == 0) {
    const float bsv = bs[0];
#pragma unroll
    for (int v = 0; v < 8; ++v) {
      float nrm = sqrtf(S1[v]);
      out[(size_t)bIdx * N_ + row0 + g * 8 + v] = S2[v] / fmaxf(nrm, 1e-12f) + bsv;
    }
  }
}

// ---------------------------------------------------------------------------------------------
extern "C" void kernel_launch(void* const* d_in, const int* in_sizes, int n_in,
                              void* d_out, int out_size, void* d_ws, size_t ws_size,
                              hipStream_t stream) {
  (void)in_sizes; (void)n_in; (void)out_size; (void)ws_size;
  const float* object_feat = (const float*)d_in[0];
  const float* lang_feat   = (const float*)d_in[1];
  const int*   lang_mask   = (const int*)d_in[2];
  const float* W1 = (const float*)d_in[3];
  const float* b1 = (const float*)d_in[4];
  const float* W2 = (const float*)d_in[5];
  const float* b2 = (const float*)d_in[6];
  const float* Wa = (const float*)d_in[7];
  const float* ba = (const float*)d_in[8];
  const float* W3 = (const float*)d_in[9];
  const float* b3 = (const float*)d_in[10];
  const float* W4 = (const float*)d_in[11];
  const float* b4 = (const float*)d_in[12];
  const float* Ws = (const float*)d_in[13];
  const float* bs = (const float*)d_in[14];
  float* out = (float*)d_out;

  // workspace layout (bytes): packed bf16 weights + lang frags + masked lang logits (~3.7 MB)
  char* ws = (char*)d_ws;
  unsigned short* W1p = (unsigned short*)(ws + 0);        // 9*16*32*32  = 147456
  unsigned short* W2p = (unsigned short*)(ws + 147456);   // 8*16*32*32  = 131072
  unsigned short* W3p = (unsigned short*)(ws + 278528);   // 147456
  unsigned short* W4p = (unsigned short*)(ws + 425984);   // 131072
  unsigned short* Lp  = (unsigned short*)(ws + 557056);   // 64*48*32*32 = 3145728
  float*          slm = (float*)(ws + 3702784);           // 64*80*4     = 20480

  pack_weight_kernel<<<18, 256, 0, stream>>>(W1, W1p, 9);
  pack_weight_kernel<<<16, 256, 0, stream>>>(W2, W2p, 8);
  pack_weight_kernel<<<18, 256, 0, stream>>>(W3, W3p, 9);
  pack_weight_kernel<<<16, 256, 0, stream>>>(W4, W4p, 8);
  pack_lang_kernel<<<384, 256, 0, stream>>>(lang_feat, Lp);
  lang_score_kernel<<<20, 256, 0, stream>>>(lang_feat, lang_mask, Wa, ba, slm);

  // 4096 tiles of 16 rows, 8 waves (one tile each) per 256-thread block
  fused_atten_kernel<<<512, 256, 0, stream>>>(object_feat, slm,
      W1p, W2p, W3p, W4p, Lp, b1, b2, b3, b4, Wa, Ws, bs, out);
}
